// Equiformerv2_68839735820494
// MI455X (gfx1250) — compile-verified, matrix operand order checked by reading, not run on hardware
//
#include <hip/hip_runtime.h>
#include <math.h>

#define NN 2048
#define EE 12288
#define CC 128
#define GG 16

typedef __attribute__((ext_vector_type(16))) _Float16 v16h;
typedef __attribute__((ext_vector_type(8)))  float    v8f;

__constant__ int c_deg[49] = {
  0,
  1,1,1,
  2,2,2,2,2,
  3,3,3,3,3,3,3,
  4,4,4,4,4,4,4,4,4,
  5,5,5,5,5,5,5,5,5,5,5,
  6,6,6,6,6,6,6,6,6,6,6,6,6};
__constant__ int c_restrict[29] = {
  0,1,2,3,4,5,6,7,8,10,11,12,13,14,18,19,20,21,22,28,29,30,31,32,40,41,42,43,44};

// ---------------- WMMA helpers (gfx1250, wave32, 16x16x32 f16 -> f32) ----------
__device__ inline v8f wmma16(v16h a, v16h b, v8f c) {
  return __builtin_amdgcn_wmma_f32_16x16x32_f16(false, a, false, b, (short)0, c,
                                                false, false);
}

// A fragment: 16x32 tile, row-major f16 source (LDS), leading dim ld (halves)
__device__ inline v16h fragA_h16(const _Float16* base, int ld) {
  int lane = threadIdx.x & 31;
  const _Float16* p = base + (lane & 15) * ld + ((lane & 16) ? 8 : 0);
  v16h f;
#pragma unroll
  for (int j = 0; j < 8; ++j) f[j] = p[j];
#pragma unroll
  for (int j = 0; j < 8; ++j) f[j + 8] = p[16 + j];
  return f;
}
// A fragment from row-major f32 source (LDS)
__device__ inline v16h fragA_f32(const float* base, int ld) {
  int lane = threadIdx.x & 31;
  const float* p = base + (lane & 15) * ld + ((lane & 16) ? 8 : 0);
  v16h f;
#pragma unroll
  for (int j = 0; j < 8; ++j) f[j] = (_Float16)p[j];
#pragma unroll
  for (int j = 0; j < 8; ++j) f[j + 8] = (_Float16)p[16 + j];
  return f;
}
// B fragment: 32x16 tile (K x N), row-major f16 source (LDS), ld = row stride
__device__ inline v16h fragB_h16(const _Float16* base, int ld) {
  int lane = threadIdx.x & 31;
  const _Float16* p = base + (lane & 15) + ((lane & 16) ? 8 : 0) * ld;
  v16h f;
#pragma unroll
  for (int j = 0; j < 8; ++j) f[j] = p[j * ld];
#pragma unroll
  for (int j = 0; j < 8; ++j) f[j + 8] = p[(16 + j) * ld];
  return f;
}
// B fragment from row-major f32 global memory
__device__ inline v16h fragB_g32(const float* base, int ld) {
  int lane = threadIdx.x & 31;
  const float* p = base + (lane & 15) + (size_t)((lane & 16) ? 8 : 0) * ld;
  v16h f;
#pragma unroll
  for (int j = 0; j < 8; ++j) f[j] = (_Float16)p[(size_t)j * ld];
#pragma unroll
  for (int j = 0; j < 8; ++j) f[j + 8] = (_Float16)p[(size_t)(16 + j) * ld];
  return f;
}

__device__ inline float siluf(float v) { return v / (1.f + expf(-v)); }

__device__ inline void atomicMaxF(float* addr, float val) {
  unsigned int* ua = (unsigned int*)addr;
  unsigned int old = *ua;
  while (__uint_as_float(old) < val) {
    unsigned int assumed = old;
    old = atomicCAS(ua, assumed, __float_as_uint(val));
    if (old == assumed) break;
  }
}

// ---------------- Generic tiled GEMM: C[M,N] = act(A[M,K]@B[K,N] + bias) ------
// 128 threads (4 waves). Block tile: 32 rows x 128 cols. M%32==0, N%128==0.
// Next K-tile is prefetched (global_prefetch_b8) while current tile computes.
template <int ACT, bool ACCUM>
__global__ __launch_bounds__(128) void k_gemm(const float* __restrict__ A,
                                              const float* __restrict__ B,
                                              const float* __restrict__ bias,
                                              float* __restrict__ C, int M,
                                              int K, int N) {
  __shared__ _Float16 As[32][40];
  __shared__ _Float16 Bs[32][136];
  const int t = threadIdx.x;
  const int wave = t >> 5;
  const int lane = t & 31;
  const int row0 = blockIdx.y << 5;
  const int col0 = blockIdx.x << 7;
  v8f acc[2][2];
#pragma unroll
  for (int a = 0; a < 2; ++a)
#pragma unroll
    for (int b = 0; b < 2; ++b)
#pragma unroll
      for (int i = 0; i < 8; ++i) acc[a][b][i] = 0.f;

  for (int k0 = 0; k0 < K; k0 += 32) {
    // prefetch next K-tile of A and B into cache while this tile is consumed
    if (k0 + 32 < K) {
      if (t < 32)  // one 128B row of next A tile per thread
        __builtin_prefetch(A + (size_t)(row0 + t) * K + (k0 + 32), 0, 1);
      // next B tile: 32 rows x 512B -> 128 cachelines, one per thread
      __builtin_prefetch(B + (size_t)(k0 + 32 + (t >> 2)) * N + col0 +
                             ((t & 3) << 5),
                         0, 1);
    }
#pragma unroll
    for (int i = 0; i < 8; ++i) {
      int idx = i * 128 + t;
      int r = idx >> 5, kk = idx & 31;
      float v = (k0 + kk < K) ? A[(size_t)(row0 + r) * K + (k0 + kk)] : 0.f;
      As[r][kk] = (_Float16)v;
    }
#pragma unroll
    for (int kk = 0; kk < 32; ++kk) {
      float v = (k0 + kk < K) ? B[(size_t)(k0 + kk) * N + (col0 + t)] : 0.f;
      Bs[kk][t] = (_Float16)v;
    }
    __syncthreads();
    const int nb = wave << 5;
    v16h af0 = fragA_h16(&As[0][0], 40);
    v16h af1 = fragA_h16(&As[16][0], 40);
#pragma unroll
    for (int nt = 0; nt < 2; ++nt) {
      v16h bf = fragB_h16(&Bs[0][nb + nt * 16], 136);
      acc[0][nt] = wmma16(af0, bf, acc[0][nt]);
      acc[1][nt] = wmma16(af1, bf, acc[1][nt]);
    }
    __syncthreads();
  }
  const int nb = wave << 5;
#pragma unroll
  for (int mt = 0; mt < 2; ++mt)
#pragma unroll
    for (int nt = 0; nt < 2; ++nt) {
      int col = col0 + nb + nt * 16 + (lane & 15);
      int rbase = row0 + mt * 16 + ((lane >> 4) << 3);
      float bv = bias ? bias[col] : 0.f;
#pragma unroll
      for (int v = 0; v < 8; ++v) {
        float val = acc[mt][nt][v] + bv;
        if (ACT == 1) val = siluf(val);
        size_t off = (size_t)(rbase + v) * N + col;
        if (ACCUM)
          C[off] += val;
        else
          C[off] = val;
      }
    }
}

// ---------------- edge geometry: gaussian smearing -----------------------------
__global__ void k_edge_geom(const float* __restrict__ ev,
                            float* __restrict__ dist) {
  int e = blockIdx.x;
  float vx = ev[e * 3 + 0], vy = ev[e * 3 + 1], vz = ev[e * 3 + 2];
  float d = sqrtf(vx * vx + vy * vy + vz * vz);
  const float delta = 5.0f / 599.0f;
  const float gco = -0.5f / ((2.f * delta) * (2.f * delta));
  for (int b = threadIdx.x; b < 600; b += blockDim.x) {
    float df = d - delta * (float)b;
    dist[(size_t)e * 600 + b] = expf(gco * df * df);
  }
}

// ---------------- real spherical harmonics (L_MAX=6) ---------------------------
__global__ void k_sph(const float* __restrict__ ev, float* __restrict__ Y) {
  int e = blockIdx.x * blockDim.x + threadIdx.x;
  if (e >= EE) return;
  float vx = ev[e * 3 + 0], vy = ev[e * 3 + 1], vz = ev[e * 3 + 2];
  float r = sqrtf(vx * vx + vy * vy + vz * vz);
  float inv = 1.0f / fmaxf(r, 1e-8f);
  float ux = vx * inv, uy = vy * inv, uz = vz * inv;
  float ct = fminf(fmaxf(uz, -1.f), 1.f);
  float st = sqrtf(fminf(fmaxf(1.f - ct * ct, 1e-12f), 1.f));
  float phi = atan2f(uy, ux);
  float P[7][7];
  P[0][0] = 1.f;
  for (int m = 1; m <= 6; ++m) P[m][m] = -(2.f * m - 1.f) * st * P[m - 1][m - 1];
  for (int m = 0; m <= 5; ++m) P[m + 1][m] = (2.f * m + 1.f) * ct * P[m][m];
  for (int m = 0; m <= 6; ++m)
    for (int l = m + 2; l <= 6; ++l)
      P[l][m] = ((2.f * l - 1.f) * ct * P[l - 1][m] -
                 (float)(l + m - 1) * P[l - 2][m]) /
                (float)(l - m);
  float* out = Y + (size_t)e * 49;
  int o = 0;
  const float FOURPI = 12.566370614359172f;
  for (int l = 0; l <= 6; ++l)
    for (int m = -l; m <= l; ++m) {
      int am = m < 0 ? -m : m;
      float nn = (2.f * l + 1.f) / FOURPI;
      for (int i = l - am + 1; i <= l + am; ++i) nn /= (float)i;
      float nrm = sqrtf(nn);
      float cval;
      if (m == 0)
        cval = nrm * P[l][0];
      else if (m > 0)
        cval = 1.41421356f * nrm * P[l][m] * cosf((float)m * phi);
      else
        cval = 1.41421356f * nrm * P[l][am] * sinf((float)am * phi);
      out[o++] = cval;
    }
}

// ---------------- embedding init ----------------------------------------------
__global__ void k_embed(const float* __restrict__ pos,
                        const float* __restrict__ et, float* __restrict__ x) {
  int n = blockIdx.x, c = threadIdx.x;
  const float LO = -3.26267f, HI = 3.295396f;
  float emb = 0.f;
  for (int j = 0; j < 3; ++j) {
    float p = pos[n * 3 + j];
    float v = (p - LO) / (HI - LO) * 128.f - 0.5f;
    int ti = (int)rintf(v);
    ti = ti < 0 ? 0 : (ti > 127 ? 127 : ti);
    emb += et[ti * 128 + c];
  }
  float* xr = x + (size_t)n * 49 * 128;
  xr[c] = emb;
  for (int k = 1; k < 49; ++k) xr[k * 128 + c] = 0.f;
}

// ---------------- x += segment_sum(Y * rad)/AVG_DEG ----------------------------
__global__ void k_scatter_init(const float* __restrict__ Y,
                               const float* __restrict__ rad896,
                               const int* __restrict__ dstI,
                               float* __restrict__ x) {
  int e = blockIdx.x, c = threadIdx.x;
  int d = dstI[e];
  for (int k = 0; k < 49; ++k) {
    float v = Y[(size_t)e * 49 + k] *
              rad896[((size_t)e * 7 + c_deg[k]) * 128 + c] * (1.0f / 3.0f);
    atomicAdd(&x[((size_t)d * 49 + k) * 128 + c], v);
  }
}

// ---------------- RMS norm over SH degrees -------------------------------------
__global__ void k_rmsnorm(const float* __restrict__ x,
                          const float* __restrict__ w, float* __restrict__ h) {
  __shared__ float red[7][128];
  __shared__ float inv[7];
  int n = blockIdx.x, c = threadIdx.x;
  float acc[7] = {0.f, 0.f, 0.f, 0.f, 0.f, 0.f, 0.f};
  const float* xn = x + (size_t)n * 49 * 128;
  for (int k = 0; k < 49; ++k) {
    float v = xn[k * 128 + c];
    acc[c_deg[k]] += v * v;
  }
  for (int l = 0; l < 7; ++l) red[l][c] = acc[l];
  __syncthreads();
  for (int off = 64; off > 0; off >>= 1) {
    if (c < off)
      for (int l = 0; l < 7; ++l) red[l][c] += red[l][c + off];
    __syncthreads();
  }
  if (c < 7) inv[c] = rsqrtf(red[c][0] / ((2.f * c + 1.f) * 128.f) + 1e-6f);
  __syncthreads();
  float* hn = h + (size_t)n * 49 * 128;
  for (int k = 0; k < 49; ++k) {
    int l = c_deg[k];
    hn[k * 128 + c] = xn[k * 128 + c] * inv[l] * w[l * 128 + c];
  }
}

// ---------------- gather l=0 rows for src/dst ----------------------------------
__global__ void k_gather0(const float* __restrict__ h,
                          const int* __restrict__ srcI,
                          const int* __restrict__ dstI, float* __restrict__ Gs,
                          float* __restrict__ Gd) {
  int idx = blockIdx.x * blockDim.x + threadIdx.x;
  int e = idx >> 7, c = idx & 127;
  Gs[idx] = h[(size_t)srcI[e] * 49 * 128 + c];
  Gd[idx] = h[(size_t)dstI[e] * 49 * 128 + c];
}

// ---------------- logits + segment max -----------------------------------------
__global__ void k_logits(const float* __restrict__ m0,
                         const float* __restrict__ radA,
                         const float* __restrict__ avec,
                         const int* __restrict__ dstI,
                         float* __restrict__ logits,
                         float* __restrict__ segmax) {
  __shared__ float red[128];
  int e = blockIdx.x, t = threadIdx.x;
  float m = m0[(size_t)e * 128 + t] * radA[(size_t)e * 128 + t];
  red[t] = siluf(m) * avec[t];
  __syncthreads();
  for (int off = 8; off > 0; off >>= 1) {
    if ((t & 15) < off) red[t] += red[t + off];
    __syncthreads();
  }
  if ((t & 15) == 0) {
    int hh = t >> 4;
    float l = red[t];
    logits[(size_t)e * 8 + hh] = l;
    atomicMaxF(&segmax[(size_t)dstI[e] * 8 + hh], l);
  }
}

__global__ void k_ex(float* __restrict__ logits,
                     const float* __restrict__ segmax,
                     const int* __restrict__ dstI, float* __restrict__ segden) {
  int idx = blockIdx.x * blockDim.x + threadIdx.x;
  int hh = idx & 7, e = idx >> 3;
  int d = dstI[e];
  float exv = expf(logits[idx] - segmax[(size_t)d * 8 + hh]);
  logits[idx] = exv;
  atomicAdd(&segden[(size_t)d * 8 + hh], exv);
}

// ---------------- fused per-edge msg/value kernel (WMMA heavy) -----------------
__global__ __launch_bounds__(128) void k_edge_msg(
    const float* __restrict__ h, const float* __restrict__ radA,
    const float* __restrict__ ws_, const float* __restrict__ wt_,
    const float* __restrict__ wv_, const float* __restrict__ ex,
    const float* __restrict__ den, const int* __restrict__ srcI,
    const int* __restrict__ dstI, float* __restrict__ agg) {
  __shared__ _Float16 hs[32][128];
  __shared__ _Float16 hd[32][128];
  __shared__ float msg[32][132];
  int e = blockIdx.x;
  int t = threadIdx.x, wave = t >> 5, lane = t & 31;
  int s = srcI[e], d = dstI[e];
  // prefetch w_v (used by second GEMM phase) while msg phase runs
  __builtin_prefetch(wv_ + (size_t)t * 128, 0, 1);
  __builtin_prefetch(wv_ + (size_t)t * 128 + 64, 0, 1);
  for (int r = 0; r < 32; ++r) {
    float vs = 0.f, vd = 0.f;
    if (r < 29) {
      int k = c_restrict[r];
      vs = h[((size_t)s * 49 + k) * 128 + t];
      vd = h[((size_t)d * 49 + k) * 128 + t];
    }
    hs[r][t] = (_Float16)vs;
    hd[r][t] = (_Float16)vd;
  }
  __syncthreads();
  const int nb = wave << 5;
  v8f acc[2][2];
#pragma unroll
  for (int a = 0; a < 2; ++a)
#pragma unroll
    for (int b = 0; b < 2; ++b)
#pragma unroll
      for (int i = 0; i < 8; ++i) acc[a][b][i] = 0.f;

  // msg = h_src_r @ w_s + h_dst_r @ w_t
  for (int k0 = 0; k0 < 128; k0 += 32) {
    v16h a0 = fragA_h16(&hs[0][k0], 128);
    v16h a1 = fragA_h16(&hs[16][k0], 128);
#pragma unroll
    for (int nt = 0; nt < 2; ++nt) {
      v16h bf = fragB_g32(ws_ + (size_t)k0 * 128 + nb + nt * 16, 128);
      acc[0][nt] = wmma16(a0, bf, acc[0][nt]);
      acc[1][nt] = wmma16(a1, bf, acc[1][nt]);
    }
    a0 = fragA_h16(&hd[0][k0], 128);
    a1 = fragA_h16(&hd[16][k0], 128);
#pragma unroll
    for (int nt = 0; nt < 2; ++nt) {
      v16h bf = fragB_g32(wt_ + (size_t)k0 * 128 + nb + nt * 16, 128);
      acc[0][nt] = wmma16(a0, bf, acc[0][nt]);
      acc[1][nt] = wmma16(a1, bf, acc[1][nt]);
    }
  }
  // msg *= radial; store to LDS (f32)
#pragma unroll
  for (int mt = 0; mt < 2; ++mt)
#pragma unroll
    for (int nt = 0; nt < 2; ++nt) {
      int col = nb + nt * 16 + (lane & 15);
      float ra = radA[(size_t)e * 128 + col];
      int rbase = mt * 16 + ((lane >> 4) << 3);
#pragma unroll
      for (int v = 0; v < 8; ++v) msg[rbase + v][col] = acc[mt][nt][v] * ra;
    }
  __syncthreads();
  // v = msg @ w_v
  v8f vac[2][2];
#pragma unroll
  for (int a = 0; a < 2; ++a)
#pragma unroll
    for (int b = 0; b < 2; ++b)
#pragma unroll
      for (int i = 0; i < 8; ++i) vac[a][b][i] = 0.f;
  for (int k0 = 0; k0 < 128; k0 += 32) {
    v16h a0 = fragA_f32(&msg[0][k0], 132);
    v16h a1 = fragA_f32(&msg[16][k0], 132);
#pragma unroll
    for (int nt = 0; nt < 2; ++nt) {
      v16h bf = fragB_g32(wv_ + (size_t)k0 * 128 + nb + nt * 16, 128);
      vac[0][nt] = wmma16(a0, bf, vac[0][nt]);
      vac[1][nt] = wmma16(a1, bf, vac[1][nt]);
    }
  }
  // scale by alpha (per head == per 16-col tile), scatter into agg[dst]
#pragma unroll
  for (int mt = 0; mt < 2; ++mt)
#pragma unroll
    for (int nt = 0; nt < 2; ++nt) {
      int head = (wave << 1) + nt;
      float alpha = ex[(size_t)e * 8 + head] /
                    fmaxf(den[(size_t)d * 8 + head], 1e-9f);
      int col = nb + nt * 16 + (lane & 15);
      int rbase = mt * 16 + ((lane >> 4) << 3);
#pragma unroll
      for (int v = 0; v < 8; ++v) {
        int row = rbase + v;
        if (row < 29)
          atomicAdd(&agg[((size_t)d * 29 + row) * 128 + col],
                    vac[mt][nt][v] * alpha);
      }
    }
}

// ---------------- FFN gating ----------------------------------------------------
__global__ void k_gate(float* __restrict__ hid) {
  float* hn = hid + (size_t)blockIdx.x * 49 * 512;
  int f = threadIdx.x;
  float s = hn[f];
  float sg = 1.f / (1.f + expf(-s));
  hn[f] = s * sg;
  for (int k = 1; k < 49; ++k) hn[k * 512 + f] *= sg;
}

// ---------------- scatters ------------------------------------------------------
__global__ void k_scatter_attn(const float* __restrict__ proj,
                               float* __restrict__ x) {
  int idx = blockIdx.x * blockDim.x + threadIdx.x;
  int c = idx & 127;
  int rr = idx >> 7;
  int r = rr % 29, n = rr / 29;
  x[((size_t)n * 49 + c_restrict[r]) * 128 + c] += proj[idx];
}

__global__ void k_scatter_latent(const float* __restrict__ proj,
                                 const int* __restrict__ batch,
                                 const float* __restrict__ cnt,
                                 float* __restrict__ out) {
  int idx = blockIdx.x * blockDim.x + threadIdx.x;
  int c = idx & 127;
  int rr = idx >> 7;
  int r = rr % 29, n = rr / 29;
  int g = batch[n];
  float invc = 1.f / fmaxf(cnt[g], 1.f);
  atomicAdd(&out[((size_t)g * 49 + c_restrict[r]) * 128 + c], proj[idx] * invc);
}

__global__ void k_fill(float* __restrict__ p, size_t n, float v) {
  size_t i = (size_t)blockIdx.x * blockDim.x + threadIdx.x;
  size_t stride = (size_t)gridDim.x * blockDim.x;
  for (; i < n; i += stride) p[i] = v;
}

__global__ void k_counts(const int* __restrict__ batch, float* __restrict__ cnt) {
  int n = blockIdx.x * blockDim.x + threadIdx.x;
  if (n < NN) atomicAdd(&cnt[batch[n]], 1.f);
}

// ---------------- host-side attention macro ------------------------------------
static void run_attention(hipStream_t stream, const float* h, const float* dist,
                          const int* srcI, const int* dstI, const float* w_s,
                          const float* w_t, const float* rw1, const float* rb1,
                          const float* rw2, const float* avec, const float* w_v,
                          const float* w_p, float* t1, float* t2, float* radA,
                          float* m0, float* logits, float* segmax,
                          float* segden, float* agg, float* proj, float* target,
                          const int* batch, const float* cnt, bool latent) {
  k_gemm<1, false><<<dim3(1, EE / 32), 128, 0, stream>>>(dist, rw1, rb1, t1, EE, 600, 128);
  k_gemm<0, false><<<dim3(1, EE / 32), 128, 0, stream>>>(t1, rw2, nullptr, radA, EE, 128, 128);
  k_gather0<<<EE * 128 / 256, 256, 0, stream>>>(h, srcI, dstI, t1, t2);
  k_gemm<0, false><<<dim3(1, EE / 32), 128, 0, stream>>>(t1, w_s, nullptr, m0, EE, 128, 128);
  k_gemm<0, true><<<dim3(1, EE / 32), 128, 0, stream>>>(t2, w_t, nullptr, m0, EE, 128, 128);
  k_fill<<<64, 256, 0, stream>>>(segmax, (size_t)NN * 8, -3.0e38f);
  k_fill<<<64, 256, 0, stream>>>(segden, (size_t)NN * 8, 0.f);
  k_logits<<<EE, 128, 0, stream>>>(m0, radA, avec, dstI, logits, segmax);
  k_ex<<<EE * 8 / 256, 256, 0, stream>>>(logits, segmax, dstI, segden);
  k_fill<<<8192, 256, 0, stream>>>(agg, (size_t)NN * 29 * 128, 0.f);
  k_edge_msg<<<EE, 128, 0, stream>>>(h, radA, w_s, w_t, w_v, logits, segden,
                                     srcI, dstI, agg);
  k_gemm<0, false><<<dim3(1, (NN * 29) / 32), 128, 0, stream>>>(
      agg, w_p, nullptr, proj, NN * 29, 128, 128);
  if (!latent)
    k_scatter_attn<<<(NN * 29 * 128) / 256, 256, 0, stream>>>(proj, target);
  else
    k_scatter_latent<<<(NN * 29 * 128) / 256, 256, 0, stream>>>(proj, batch,
                                                                cnt, target);
}

extern "C" void kernel_launch(void* const* d_in, const int* in_sizes, int n_in,
                              void* d_out, int out_size, void* d_ws,
                              size_t ws_size, hipStream_t stream) {
  (void)in_sizes; (void)n_in; (void)out_size; (void)ws_size;
  const float* pos         = (const float*)d_in[0];
  const float* edge_vec    = (const float*)d_in[1];
  const int*   edge_index  = (const int*)d_in[2];
  const int*   batch       = (const int*)d_in[3];
  const float* embed_table = (const float*)d_in[4];
  const float* deg_w1      = (const float*)d_in[5];
  const float* deg_b1      = (const float*)d_in[6];
  const float* deg_w2      = (const float*)d_in[7];
  const float* deg_b2      = (const float*)d_in[8];
  const float* deg_w3      = (const float*)d_in[9];
  const float* attn_norm_w = (const float*)d_in[10];
  const float* w_src       = (const float*)d_in[11];
  const float* w_tgt       = (const float*)d_in[12];
  const float* rad_w1      = (const float*)d_in[13];
  const float* rad_b1      = (const float*)d_in[14];
  const float* rad_w2      = (const float*)d_in[15];
  const float* alpha_vec   = (const float*)d_in[16];
  const float* w_val       = (const float*)d_in[17];
  const float* w_proj      = (const float*)d_in[18];
  const float* ffn_norm_w  = (const float*)d_in[19];
  const float* ffn_w1      = (const float*)d_in[20];
  const float* ffn_w2      = (const float*)d_in[21];
  const float* final_norm_w= (const float*)d_in[22];
  const float* lat_w_src   = (const float*)d_in[23];
  const float* lat_w_tgt   = (const float*)d_in[24];
  const float* lat_rad_w1  = (const float*)d_in[25];
  const float* lat_rad_b1  = (const float*)d_in[26];
  const float* lat_rad_w2  = (const float*)d_in[27];
  const float* lat_alpha   = (const float*)d_in[28];
  const float* lat_w_val   = (const float*)d_in[29];
  const float* lat_w_proj  = (const float*)d_in[30];

  const int* srcI = edge_index;
  const int* dstI = edge_index + EE;

  float* W = (float*)d_ws;
  float* dist   = W;                                 // E*600
  float* x      = dist + (size_t)EE * 600;           // N*49*128
  float* hbuf   = x + (size_t)NN * 49 * 128;         // N*49*128
  float* Ybuf   = hbuf + (size_t)NN * 49 * 128;      // E*49
  float* t1     = Ybuf + (size_t)EE * 49;            // E*128
  float* t2     = t1 + (size_t)EE * 128;             // E*128
  float* radA   = t2 + (size_t)EE * 128;             // E*128
  float* m0     = radA + (size_t)EE * 128;           // E*128
  float* logits = m0 + (size_t)EE * 128;             // E*8
  float* segmax = logits + (size_t)EE * 8;           // N*8
  float* segden = segmax + (size_t)NN * 8;           // N*8
  float* agg    = segden + (size_t)NN * 8;           // N*29*128
  float* big    = agg + (size_t)NN * 29 * 128;       // E*7*128 (reused)
  float* cnt    = big + (size_t)EE * 7 * 128;        // G

  // 1) edge geometry + spherical harmonics + embedding
  k_edge_geom<<<EE, 128, 0, stream>>>(edge_vec, dist);
  k_sph<<<(EE + 127) / 128, 128, 0, stream>>>(edge_vec, Ybuf);
  k_embed<<<NN, 128, 0, stream>>>(pos, embed_table, x);

  // 2) degree-embedding MLP over edges, then scatter Y*rad into x
  k_gemm<1, false><<<dim3(1, EE / 32), 128, 0, stream>>>(dist, deg_w1, deg_b1, t1, EE, 600, 128);
  k_gemm<1, false><<<dim3(1, EE / 32), 128, 0, stream>>>(t1, deg_w2, deg_b2, t2, EE, 128, 128);
  k_gemm<0, false><<<dim3(7, EE / 32), 128, 0, stream>>>(t2, deg_w3, nullptr, big, EE, 128, 896);
  k_scatter_init<<<EE, 128, 0, stream>>>(Ybuf, big, dstI, x);

  // 3) transformer layers
  for (int i = 0; i < 2; ++i) {
    k_rmsnorm<<<NN, 128, 0, stream>>>(x, attn_norm_w + (size_t)i * 7 * 128, hbuf);
    run_attention(stream, hbuf, dist, srcI, dstI,
                  w_src + (size_t)i * 128 * 128, w_tgt + (size_t)i * 128 * 128,
                  rad_w1 + (size_t)i * 600 * 128, rad_b1 + (size_t)i * 128,
                  rad_w2 + (size_t)i * 128 * 128, alpha_vec + (size_t)i * 128,
                  w_val + (size_t)i * 128 * 128, w_proj + (size_t)i * 128 * 128,
                  t1, t2, radA, m0, logits, segmax, segden, agg, big, x,
                  nullptr, nullptr, false);
    k_rmsnorm<<<NN, 128, 0, stream>>>(x, ffn_norm_w + (size_t)i * 7 * 128, hbuf);
    for (int c = 0; c < 8; ++c) {  // FFN in node chunks of 256 (hid in `big`)
      size_t base = (size_t)c * 256 * 49 * 128;
      k_gemm<0, false><<<dim3(4, 392), 128, 0, stream>>>(
          hbuf + base, ffn_w1 + (size_t)i * 128 * 512, nullptr, big, 12544, 128, 512);
      k_gate<<<256, 512, 0, stream>>>(big);
      k_gemm<0, true><<<dim3(1, 392), 128, 0, stream>>>(
          big, ffn_w2 + (size_t)i * 512 * 128, nullptr, x + base, 12544, 512, 128);
    }
  }

  // 4) final norm + latent attention pooled straight into d_out
  k_rmsnorm<<<NN, 128, 0, stream>>>(x, final_norm_w, hbuf);
  k_fill<<<1, 64, 0, stream>>>(cnt, (size_t)GG, 0.f);
  k_counts<<<8, 256, 0, stream>>>(batch, cnt);
  k_fill<<<392, 256, 0, stream>>>((float*)d_out, (size_t)GG * 49 * 128, 0.f);
  run_attention(stream, hbuf, dist, srcI, dstI, lat_w_src, lat_w_tgt,
                lat_rad_w1, lat_rad_b1, lat_rad_w2, lat_alpha, lat_w_val,
                lat_w_proj, t1, t2, radA, m0, logits, segmax, segden, agg, big,
                (float*)d_out, batch, cnt, true);
}
